// TemporalGCN_14285061226985
// MI455X (gfx1250) — matrix operand order repaired and checked by reading, not compiled, not run on hardware
//
#include <hip/hip_runtime.h>
#include <hip/hip_bf16.h>

typedef __attribute__((ext_vector_type(16))) __bf16 v16bf;
typedef __attribute__((ext_vector_type(8)))  float  v8f;

#define T_STEPS 20
#define NPS     400
#define B_SZ    128
#define NNODE   (B_SZ * NPS)     // 51200
#define E_CNT   819200
#define DIN     128
#define HDIM    64

// ---------------------------------------------------------------- utilities
__global__ void fill_f32(float* p, float v, long long n) {
    long long i = (long long)blockIdx.x * blockDim.x + threadIdx.x;
    if (i < n) p[i] = v;
}

__global__ void rsqrt_kernel(float* p, long long n) {
    long long i = (long long)blockIdx.x * blockDim.x + threadIdx.x;
    if (i < n) { float d = p[i]; p[i] = (d > 0.0f) ? rsqrtf(d) : 0.0f; }
}

// Wt[h*K + k] = (bf16)W[k*H + h]   (B-fragment friendly: contiguous K per col)
__global__ void conv_transpose_bf16(const float* __restrict__ W, __bf16* __restrict__ Wt,
                                    int K, int H) {
    int i = blockIdx.x * blockDim.x + threadIdx.x;
    if (i < K * H) { int k = i / H, h = i % H; Wt[h * K + k] = (__bf16)W[i]; }
}

__global__ void conv_bf16(const float* __restrict__ W, __bf16* __restrict__ O, int n) {
    int i = blockIdx.x * blockDim.x + threadIdx.x;
    if (i < n) O[i] = (__bf16)W[i];
}

__global__ void add_vec(const float* a, const float* b, float* o, int n) {
    int i = blockIdx.x * blockDim.x + threadIdx.x;
    if (i < n) o[i] = a[i] + b[i];
}

// ------------------------------------------------------------- degree count
__global__ void deg_kernel(const long long* __restrict__ ei, float* __restrict__ deg) {
    long long tid = (long long)blockIdx.x * blockDim.x + threadIdx.x;
    if (tid >= (long long)T_STEPS * E_CNT) return;
    int t = (int)(tid / E_CNT);
    long long e = tid % E_CNT;
    int d = (int)ei[((size_t)t * 2 + 1) * E_CNT + e];
    atomicAdd(deg + (size_t)t * NNODE + d, 1.0f);
}

// ------------------------------------------------------- WMMA bf16 fragments
__device__ __forceinline__ v16bf load_bfrag(const __bf16* base) {
    v16bf b;
#pragma unroll
    for (int j = 0; j < 16; ++j) b[j] = base[j];
    return b;
}

// C[64-row tile][64] = A[rows][KDIM] @ Wt^T, A masked (per-node ego) and cast to bf16.
// 128 threads = 4 waves; each wave owns a 16-row M tile, all 4 N tiles.
template <int KDIM, bool MASK>
__global__ __launch_bounds__(128)
void gemm_wmma_kernel(const float* __restrict__ A,
                      const unsigned char* __restrict__ ego, int t,
                      const __bf16* __restrict__ Wt,    // [64][KDIM]
                      float* __restrict__ Out)          // [rows][64]
{
    const int lane = threadIdx.x & 31;
    const int wave = threadIdx.x >> 5;
    const int half = lane >> 4;
    const int col  = lane & 15;
    const long long rowBase = (long long)blockIdx.x * 64;
    const long long mrow    = rowBase + wave * 16 + (lane & 15);

    float mask = 1.0f;
    if constexpr (MASK) {
        int b = (int)(mrow / NPS), p = (int)(mrow % NPS);
        mask = ego[((size_t)b * T_STEPS + t) * NPS + p] ? 1.0f : 0.0f;
    }

    constexpr int NCH = KDIM / 32;
    const float* arow = A + (size_t)mrow * KDIM;
    v16bf afrag[NCH];
#pragma unroll
    for (int c = 0; c < NCH; ++c) {
#pragma unroll
        for (int j = 0; j < 8; ++j)
            afrag[c][j]     = (__bf16)(arow[c * 32 + half * 16 + j] * mask);
#pragma unroll
        for (int j = 0; j < 8; ++j)
            afrag[c][8 + j] = (__bf16)(arow[c * 32 + half * 16 + 8 + j] * mask);
    }

#pragma unroll
    for (int nt = 0; nt < 4; ++nt) {
        v8f acc = {};
#pragma unroll
        for (int c = 0; c < NCH; ++c) {
            v16bf bfrag = load_bfrag(Wt + (size_t)(nt * 16 + col) * KDIM + c * 32 + half * 16);
            acc = __builtin_amdgcn_wmma_f32_16x16x32_bf16(
                false, afrag[c], false, bfrag, (short)0, acc, false, false);
        }
#pragma unroll
        for (int r = 0; r < 8; ++r) {
            long long gr = rowBase + wave * 16 + r + 8 * half;
            Out[(size_t)gr * HDIM + nt * 16 + col] = acc[r];
        }
    }
}

// -------------------------------------------------------- edge scatter-add
// one edge per 16 threads; each thread handles 4 feature channels
__global__ void agg_kernel(const long long* __restrict__ eit,  // [2][E]
                           const float* __restrict__ dinv_t,   // [N]
                           const float* __restrict__ Xw,       // [N][64]
                           float* __restrict__ Acc)            // [N][64]
{
    long long tid = (long long)blockIdx.x * blockDim.x + threadIdx.x;
    long long e   = tid >> 4;
    int q         = (int)(tid & 15);
    if (e >= E_CNT) return;
    int s = (int)eit[e];
    int d = (int)eit[E_CNT + e];
    float norm = dinv_t[s] * dinv_t[d];
    float4 v = *(const float4*)(Xw + (size_t)s * HDIM + q * 4);
    float* ap = Acc + (size_t)d * HDIM + q * 4;
    atomicAdd(ap + 0, norm * v.x);
    atomicAdd(ap + 1, norm * v.y);
    atomicAdd(ap + 2, norm * v.z);
    atomicAdd(ap + 3, norm * v.w);
}

// ------------------------------------------------------- layer finalizers
__global__ void fin1_kernel(const float* __restrict__ Acc, const float* __restrict__ Xw,
                            const float* __restrict__ dinv_t, const float* __restrict__ b1,
                            float* __restrict__ H1) {
    long long i = (long long)blockIdx.x * blockDim.x + threadIdx.x;
    if (i >= (long long)NNODE * HDIM) return;
    long long n = i >> 6; int h = (int)(i & 63);
    float di = dinv_t[n];
    float v = Acc[i] + di * di * Xw[i] + b1[h];
    H1[i] = fmaxf(v, 0.0f);
}

__global__ void fin2_kernel(const float* __restrict__ Acc, const float* __restrict__ Xw,
                            const float* __restrict__ dinv_t, const float* __restrict__ b2,
                            const unsigned char* __restrict__ ego, int t,
                            __bf16* __restrict__ SeqT) {
    long long i = (long long)blockIdx.x * blockDim.x + threadIdx.x;
    if (i >= (long long)NNODE * HDIM) return;
    long long n = i >> 6; int h = (int)(i & 63);
    float di = dinv_t[n];
    float v = Acc[i] + di * di * Xw[i] + b2[h];
    int b = (int)(n / NPS), p = (int)(n % NPS);
    float neg = ego[((size_t)b * T_STEPS + t) * NPS + p] ? 0.0f : 1.0f;
    SeqT[i] = (__bf16)(v * neg);
}

// ---------------------------------------------------------- persistent LSTM
// 800 blocks x 128 threads; wave owns 16 node rows; h,c live in registers
// across all 20 timesteps. z = [x h] @ [Wih; Whh]^T computed with 64 WMMAs
// per step per wave (16 gate-col tiles x (2 K-chunks x 2 weight mats)).
__global__ __launch_bounds__(128)
void lstm_kernel(const __bf16* __restrict__ seq,   // [T][N][64]
                 const __bf16* __restrict__ wihb,  // [256][64] bf16
                 const __bf16* __restrict__ whhb,  // [256][64] bf16
                 const float* __restrict__ bsum,   // [256]
                 float* __restrict__ hn)           // [N][64]
{
    __shared__ __bf16 lds_h[64 * 64];   // h state staged D-layout -> A-layout

    const int lane = threadIdx.x & 31;
    const int wave = threadIdx.x >> 5;
    const int half = lane >> 4;
    const int col  = lane & 15;
    const int row  = lane & 15;
    const long long nodeBase = (long long)blockIdx.x * 64;

    float hst[4][8] = {};
    float cst[4][8] = {};

    for (int t = 0; t < T_STEPS; ++t) {
        __syncthreads();   // previous iter's fragment reads complete
#pragma unroll
        for (int q = 0; q < 4; ++q)
#pragma unroll
            for (int r = 0; r < 8; ++r)
                lds_h[(wave * 16 + r + 8 * half) * 64 + q * 16 + col] = (__bf16)hst[q][r];
        __syncthreads();

        // A fragments: x from global (bf16 seq), h from LDS
        const __bf16* xrow = seq + ((size_t)t * NNODE + nodeBase + wave * 16 + row) * HDIM;
        const __bf16* hrow = &lds_h[(wave * 16 + row) * 64];
        v16bf ax[2], ah[2];
#pragma unroll
        for (int c = 0; c < 2; ++c) {
#pragma unroll
            for (int j = 0; j < 8; ++j) {
                ax[c][j]     = xrow[c * 32 + half * 16 + j];
                ax[c][8 + j] = xrow[c * 32 + half * 16 + 8 + j];
                ah[c][j]     = hrow[c * 32 + half * 16 + j];
                ah[c][8 + j] = hrow[c * 32 + half * 16 + 8 + j];
            }
        }

        float z[16][8];
#pragma unroll
        for (int nt = 0; nt < 16; ++nt) {
            v8f acc = {};
#pragma unroll
            for (int c = 0; c < 2; ++c) {
                v16bf bi = load_bfrag(wihb + (size_t)(nt * 16 + col) * 64 + c * 32 + half * 16);
                acc = __builtin_amdgcn_wmma_f32_16x16x32_bf16(
                    false, ax[c], false, bi, (short)0, acc, false, false);
                v16bf bh = load_bfrag(whhb + (size_t)(nt * 16 + col) * 64 + c * 32 + half * 16);
                acc = __builtin_amdgcn_wmma_f32_16x16x32_bf16(
                    false, ah[c], false, bh, (short)0, acc, false, false);
            }
            float bias = bsum[nt * 16 + col];
#pragma unroll
            for (int r = 0; r < 8; ++r) z[nt][r] = acc[r] + bias;
        }

        // gates: nt = gate*4 + hcol_tile, torch order i,f,g,o
#pragma unroll
        for (int q = 0; q < 4; ++q)
#pragma unroll
            for (int r = 0; r < 8; ++r) {
                float ig = 1.0f / (1.0f + __expf(-z[q][r]));
                float fg = 1.0f / (1.0f + __expf(-z[4 + q][r]));
                float gg = tanhf(z[8 + q][r]);
                float og = 1.0f / (1.0f + __expf(-z[12 + q][r]));
                float c  = fg * cst[q][r] + ig * gg;
                cst[q][r] = c;
                hst[q][r] = og * tanhf(c);
            }
    }

#pragma unroll
    for (int q = 0; q < 4; ++q)
#pragma unroll
        for (int r = 0; r < 8; ++r)
            hn[(size_t)(nodeBase + wave * 16 + r + 8 * half) * HDIM + q * 16 + col] = hst[q][r];
}

// ------------------------------------------------------------------ tiny FC
__global__ void fc_kernel(const float* __restrict__ hnv, const float* __restrict__ fcw,
                          const float* __restrict__ fcb, float* __restrict__ out) {
    long long i = (long long)blockIdx.x * blockDim.x + threadIdx.x;
    if (i >= (long long)NNODE * 2) return;
    long long n = i >> 1; int d = (int)(i & 1);
    float s = fcb[d];
    const float* hp = hnv + (size_t)n * HDIM;
    const float* wp = fcw + d * HDIM;
#pragma unroll
    for (int h = 0; h < HDIM; ++h) s += hp[h] * wp[h];
    out[i] = s;
}

// ------------------------------------------------------------------ launcher
extern "C" void kernel_launch(void* const* d_in, const int* in_sizes, int n_in,
                              void* d_out, int out_size, void* d_ws, size_t ws_size,
                              hipStream_t stream) {
    const float*         x    = (const float*)d_in[0];
    const long long*     ei   = (const long long*)d_in[1];
    const unsigned char* ego  = (const unsigned char*)d_in[2];
    const float*         w1   = (const float*)d_in[3];
    const float*         b1   = (const float*)d_in[4];
    const float*         w2   = (const float*)d_in[5];
    const float*         b2   = (const float*)d_in[6];
    const float*         wih  = (const float*)d_in[7];
    const float*         whh  = (const float*)d_in[8];
    const float*         bih  = (const float*)d_in[9];
    const float*         bhh  = (const float*)d_in[10];
    const float*         fcw  = (const float*)d_in[11];
    const float*         fcb  = (const float*)d_in[12];
    float*               out  = (float*)d_out;

    char*  ws  = (char*)d_ws;
    size_t off = 0;
    auto take = [&](size_t bytes) -> char* {
        char* p = ws + off;
        off = (off + bytes + 255) & ~(size_t)255;
        return p;
    };
    float*  dinv = (float*)take(sizeof(float) * (size_t)T_STEPS * NNODE);
    float*  xw   = (float*)take(sizeof(float) * (size_t)NNODE * HDIM);
    float*  acc  = (float*)take(sizeof(float) * (size_t)NNODE * HDIM);
    float*  h1   = (float*)take(sizeof(float) * (size_t)NNODE * HDIM);
    float*  hnb  = (float*)take(sizeof(float) * (size_t)NNODE * HDIM);
    __bf16* seq  = (__bf16*)take(sizeof(__bf16) * (size_t)T_STEPS * NNODE * HDIM);
    __bf16* w1t  = (__bf16*)take(sizeof(__bf16) * HDIM * DIN);
    __bf16* w2t  = (__bf16*)take(sizeof(__bf16) * HDIM * HDIM);
    __bf16* wihb = (__bf16*)take(sizeof(__bf16) * 4 * HDIM * HDIM);
    __bf16* whhb = (__bf16*)take(sizeof(__bf16) * 4 * HDIM * HDIM);
    float*  bsum = (float*)take(sizeof(float) * 4 * HDIM);

    // degree -> d^{-1/2} for all timesteps
    fill_f32<<<(T_STEPS * NNODE + 255) / 256, 256, 0, stream>>>(
        dinv, 1.0f, (long long)T_STEPS * NNODE);
    deg_kernel<<<(int)(((long long)T_STEPS * E_CNT + 255) / 256), 256, 0, stream>>>(ei, dinv);
    rsqrt_kernel<<<(T_STEPS * NNODE + 255) / 256, 256, 0, stream>>>(
        dinv, (long long)T_STEPS * NNODE);

    // weight prep (once): bf16, B-fragment-friendly layouts
    conv_transpose_bf16<<<(DIN * HDIM + 255) / 256, 256, 0, stream>>>(w1, w1t, DIN, HDIM);
    conv_transpose_bf16<<<(HDIM * HDIM + 255) / 256, 256, 0, stream>>>(w2, w2t, HDIM, HDIM);
    conv_bf16<<<(4 * HDIM * HDIM + 255) / 256, 256, 0, stream>>>(wih, wihb, 4 * HDIM * HDIM);
    conv_bf16<<<(4 * HDIM * HDIM + 255) / 256, 256, 0, stream>>>(whh, whhb, 4 * HDIM * HDIM);
    add_vec<<<1, 256, 0, stream>>>(bih, bhh, bsum, 4 * HDIM);

    const int nGemmBlk = NNODE / 64;                 // 800
    const int nAggBlk  = (E_CNT * 16) / 256;         // 51200
    const int nEltBlk  = (NNODE * HDIM) / 256;       // 12800

    for (int t = 0; t < T_STEPS; ++t) {
        const long long* eit    = ei + (size_t)t * 2 * E_CNT;
        const float*     dinv_t = dinv + (size_t)t * NNODE;

        fill_f32<<<nEltBlk, 256, 0, stream>>>(acc, 0.0f, (long long)NNODE * HDIM);
        gemm_wmma_kernel<DIN, true><<<nGemmBlk, 128, 0, stream>>>(
            x + (size_t)t * NNODE * DIN, ego, t, w1t, xw);
        agg_kernel<<<nAggBlk, 256, 0, stream>>>(eit, dinv_t, xw, acc);
        fin1_kernel<<<nEltBlk, 256, 0, stream>>>(acc, xw, dinv_t, b1, h1);

        fill_f32<<<nEltBlk, 256, 0, stream>>>(acc, 0.0f, (long long)NNODE * HDIM);
        gemm_wmma_kernel<HDIM, false><<<nGemmBlk, 128, 0, stream>>>(
            h1, (const unsigned char*)nullptr, t, w2t, xw);
        agg_kernel<<<nAggBlk, 256, 0, stream>>>(eit, dinv_t, xw, acc);
        fin2_kernel<<<nEltBlk, 256, 0, stream>>>(
            acc, xw, dinv_t, b2, ego, t, seq + (size_t)t * NNODE * HDIM);
    }

    lstm_kernel<<<NNODE / 64, 128, 0, stream>>>(seq, wihb, whhb, bsum, hnb);
    fc_kernel<<<(NNODE * 2 + 255) / 256, 256, 0, stream>>>(hnb, fcw, fcb, out);
}
